// TransMIL_fusion_13108240187923
// MI455X (gfx1250) — matrix operand order repaired and checked by reading, not compiled, not run on hardware
//
#include <hip/hip_runtime.h>

// ---------------------------------------------------------------------------
// Types / helpers
// ---------------------------------------------------------------------------
typedef __attribute__((ext_vector_type(16))) __bf16          v16bf;
typedef __attribute__((ext_vector_type(8)))  float           v8f;
typedef __attribute__((ext_vector_type(2)))  float           v2f;
typedef __attribute__((ext_vector_type(8)))  unsigned short  v8us;

union FragBF { v8us h[2]; v16bf v; };

__device__ __forceinline__ unsigned short f2bf(float f) {
  unsigned int x = __float_as_uint(f);
  unsigned int r = (x + 0x7FFFu + ((x >> 16) & 1u)) >> 16;   // RNE
  return (unsigned short)r;
}
__device__ __forceinline__ float bf2f(unsigned short u) {
  return __uint_as_float(((unsigned int)u) << 16);
}

// ---------------------------------------------------------------------------
// Elementwise convert / copy
// ---------------------------------------------------------------------------
__global__ void f32_to_bf16_k(const float* __restrict__ x, unsigned short* __restrict__ y, long long n) {
  for (long long i = (long long)blockIdx.x * 256 + threadIdx.x; i < n; i += (long long)gridDim.x * 256)
    y[i] = f2bf(x[i]);
}
__global__ void copy_f32_k(const float* __restrict__ x, float* __restrict__ y, long long n) {
  for (long long i = (long long)blockIdx.x * 256 + threadIdx.x; i < n; i += (long long)gridDim.x * 256)
    y[i] = x[i];
}

// ---------------------------------------------------------------------------
// BF16 WMMA GEMM: D = act(A(bf16) x B(bf16) + bias) [+ Cres], row-offset store,
// fp32 or bf16 output. Block = 256 thr (8 waves), tile 128x128, BK=32.
// A staged row-major (k-contiguous), B staged TRANSPOSED (n-major,
// k-contiguous) so both fragments are 16B-contiguous LDS loads per the
// CDNA5 ISA 7.12.2 wave32 layouts (A: two 16B runs; B: two 16B runs).
// ---------------------------------------------------------------------------
struct GemmBF {
  const unsigned short* A; long long lda, sA;
  const unsigned short* B; long long ldb, sB;
  int transB;
  const float* bias;
  int relu;
  float* Cf; unsigned short* Cb; long long ldc, sC;
  const float* Cres; long long ldres;
  int rowOff;
  int M, N, K;
};

#define AK 40   // padded k-stride (elements): 80B rows, 16B-aligned chunks

__global__ __launch_bounds__(256) void gemm_bf16_k(GemmBF p) {
  __shared__ unsigned short As[128][AK];   // [m][k]
  __shared__ unsigned short Bs[128][AK];   // [n][k]  (transposed stage)
  const int tid  = threadIdx.x;
  const int m0   = blockIdx.y * 128, n0 = blockIdx.x * 128;
  const long long bat = blockIdx.z;
  const unsigned short* A = p.A + bat * p.sA;
  const unsigned short* B = p.B + bat * p.sB;
  const int lane = tid & 31, wave = tid >> 5;
  const int wy = wave >> 1, wx = wave & 1;        // 4x2 waves -> 32x64 per wave
  const int half = lane >> 4, r16 = lane & 15;

  const v8f  vz  = {0.f,0.f,0.f,0.f,0.f,0.f,0.f,0.f};
  const v8us uz  = {0,0,0,0,0,0,0,0};
  v8f acc[2][4];
  for (int i = 0; i < 2; ++i) for (int j = 0; j < 4; ++j) acc[i][j] = vz;

  for (int k0 = 0; k0 < p.K; k0 += 32) {
    const bool kfull = (k0 + 32 <= p.K);
    // ---- stage A: 128 rows x 32 k, 16B vector loads (k-contiguous) ----
    for (int i = tid; i < 128 * 4; i += 256) {
      int r = i >> 2, c8 = (i & 3) * 8;
      int gr = m0 + r;
      v8us val = uz;
      if (gr < p.M) {
        if (kfull) val = *(const v8us*)&A[(long long)gr * p.lda + k0 + c8];
        else for (int j = 0; j < 8; ++j) { int gc = k0 + c8 + j; if (gc < p.K) val[j] = A[(long long)gr * p.lda + gc]; }
      }
      *(v8us*)&As[r][c8] = val;
    }
    // ---- stage B transposed into Bs[n][k] ----
    if (p.transB) {   // B stored (N,K): k-contiguous in memory -> vector loads
      for (int i = tid; i < 128 * 4; i += 256) {
        int c = i >> 2, k8 = (i & 3) * 8;
        int gn = n0 + c;
        v8us val = uz;
        if (gn < p.N) {
          if (kfull) val = *(const v8us*)&B[(long long)gn * p.ldb + k0 + k8];
          else for (int j = 0; j < 8; ++j) { int gc = k0 + k8 + j; if (gc < p.K) val[j] = B[(long long)gn * p.ldb + gc]; }
        }
        *(v8us*)&Bs[c][k8] = val;
      }
    } else {          // B stored (K,N): n-contiguous -> vector load, scatter to LDS
      for (int i = tid; i < 32 * 16; i += 256) {
        int r = i >> 4, c8 = (i & 15) * 8;
        int gk = k0 + r;
        v8us val = uz;
        if (gk < p.K) {
          if (n0 + c8 + 7 < p.N) val = *(const v8us*)&B[(long long)gk * p.ldb + n0 + c8];
          else for (int j = 0; j < 8; ++j) { int gn = n0 + c8 + j; if (gn < p.N) val[j] = B[(long long)gk * p.ldb + gn]; }
        }
        for (int j = 0; j < 8; ++j) Bs[c8 + j][r] = val[j];
      }
    }
    if (k0 + 32 < p.K) {   // stream hint for next A tile
      int pr = m0 + (tid >> 1);
      if (pr < p.M) __builtin_prefetch(&A[(long long)pr * p.lda + k0 + 32], 0, 1);
    }
    __syncthreads();

    // ---- fragments: contiguous 16B LDS loads per ISA 7.12.2 ----
    FragBF af[2], bf[4];
    for (int rt = 0; rt < 2; ++rt) {
      int row = wy * 32 + rt * 16 + r16;
      af[rt].h[0] = *(const v8us*)&As[row][half * 8];        // K = half*8 + 0..7
      af[rt].h[1] = *(const v8us*)&As[row][16 + half * 8];   // K = 16+half*8 + 0..7
    }
    for (int ct = 0; ct < 4; ++ct) {
      int col = wx * 64 + ct * 16 + r16;
      bf[ct].h[0] = *(const v8us*)&Bs[col][half * 8];        // K = half*8 + 0..7
      bf[ct].h[1] = *(const v8us*)&Bs[col][16 + half * 8];   // K = 16+half*8 + 0..7
    }
    for (int rt = 0; rt < 2; ++rt)
      for (int ct = 0; ct < 4; ++ct)
        acc[rt][ct] = __builtin_amdgcn_wmma_f32_16x16x32_bf16(
            false, af[rt].v, false, bf[ct].v, (short)0, acc[rt][ct], false, false);
    __syncthreads();
  }

  for (int rt = 0; rt < 2; ++rt)
    for (int ct = 0; ct < 4; ++ct)
      for (int i = 0; i < 8; ++i) {
        int row = m0 + wy * 32 + rt * 16 + half * 8 + i;
        int col = n0 + wx * 64 + ct * 16 + r16;
        if (row >= p.M || col >= p.N || row < p.rowOff) continue;
        float v = acc[rt][ct][i];
        if (p.bias) v += p.bias[col];
        if (p.relu) v = fmaxf(v, 0.f);
        long long orow = row - p.rowOff;
        if (p.Cres) v += p.Cres[orow * p.ldres + col];
        if (p.Cf) p.Cf[bat * p.sC + orow * p.ldc + col] = v;
        else      p.Cb[bat * p.sC + orow * p.ldc + col] = f2bf(v);
      }
}

// ---------------------------------------------------------------------------
// FP32 WMMA GEMM (V_WMMA_F32_16X16X4_F32) for the precision-sensitive
// Newton-Schulz pinv chain.  D = alpha*(A@B) + beta*Madd.
// Block = 256 thr (8 waves), tile 64x64, BK=16 (4 WMMA K-steps).
// B staged transposed so both fragments are contiguous 8B LDS loads.
// ---------------------------------------------------------------------------
struct GemmF32 {
  const float* A; long long lda, sA;
  const float* B; long long ldb, sB;
  float* C; long long ldc, sC;
  const float* Madd; long long ldm, sM;
  float alpha, beta;
  int M, N, K;
};

__global__ __launch_bounds__(256) void gemm_f32_k(GemmF32 p) {
  __shared__ float As[64][16];   // [m][k]
  __shared__ float Bs[64][18];   // [n][k] transposed, padded (72B rows, 8B-aligned)
  const int tid = threadIdx.x;
  const int m0 = blockIdx.y * 64, n0 = blockIdx.x * 64;
  const long long bat = blockIdx.z;
  const float* A = p.A + bat * p.sA;
  const float* B = p.B + bat * p.sB;
  const int lane = tid & 31, wave = tid >> 5;
  const int wy = wave >> 2, wx = wave & 3;      // 2x4 waves -> 32x16 per wave
  const int half = lane >> 4, r16 = lane & 15;

  const v8f vz = {0.f,0.f,0.f,0.f,0.f,0.f,0.f,0.f};
  v8f acc[2]; acc[0] = vz; acc[1] = vz;

  for (int k0 = 0; k0 < p.K; k0 += 16) {
    for (int i = tid; i < 64 * 16; i += 256) {
      int r = i >> 4, c = i & 15;
      int gr = m0 + r, gc = k0 + c;
      As[r][c] = (gr < p.M && gc < p.K) ? A[(long long)gr * p.lda + gc] : 0.f;
    }
    for (int i = tid; i < 16 * 64; i += 256) {
      int r = i >> 6, c = i & 63;       // r = k, c = n
      int gk = k0 + r, gn = n0 + c;
      Bs[c][r] = (gk < p.K && gn < p.N) ? B[(long long)gk * p.ldb + gn] : 0.f;
    }
    __syncthreads();
    for (int kk = 0; kk < 4; ++kk) {
      int kb = kk * 4;
      int col = wx * 16 + r16;
      v2f b = *(const v2f*)&Bs[col][kb + half * 2];   // K = kb + half*2, +1
      for (int rt = 0; rt < 2; ++rt) {
        int row = wy * 32 + rt * 16 + r16;
        v2f a = *(const v2f*)&As[row][kb + half * 2];
        acc[rt] = __builtin_amdgcn_wmma_f32_16x16x4_f32(
            false, a, false, b, (short)0, acc[rt], false, false);
      }
    }
    __syncthreads();
  }

  for (int rt = 0; rt < 2; ++rt)
    for (int i = 0; i < 8; ++i) {
      int row = m0 + wy * 32 + rt * 16 + half * 8 + i;
      int col = n0 + wx * 16 + r16;
      if (row >= p.M || col >= p.N) continue;
      float v = p.alpha * acc[rt][i];
      if (p.Madd) v += p.beta * p.Madd[bat * p.sM + (long long)row * p.ldm + col];
      p.C[bat * p.sC + (long long)row * p.ldc + col] = v;
    }
}

// ---------------------------------------------------------------------------
// LayerNorm over 512-wide rows, bf16 output (feeds qkv GEMM)
// ---------------------------------------------------------------------------
__global__ __launch_bounds__(256) void layernorm_bf16_k(const float* __restrict__ X,
    const float* __restrict__ g, const float* __restrict__ b, unsigned short* __restrict__ Y) {
  const int row = blockIdx.x, tid = threadIdx.x;
  const float* x = X + (long long)row * 512;
  float x0 = x[tid], x1 = x[tid + 256];
  __shared__ float s1[256], s2[256];
  s1[tid] = x0 + x1; s2[tid] = x0 * x0 + x1 * x1;
  __syncthreads();
  for (int st = 128; st > 0; st >>= 1) {
    if (tid < st) { s1[tid] += s1[tid + st]; s2[tid] += s2[tid + st]; }
    __syncthreads();
  }
  float mean = s1[0] * (1.f / 512.f);
  float var  = s2[0] * (1.f / 512.f) - mean * mean;
  float inv  = rsqrtf(var + 1e-5f);
  unsigned short* y = Y + (long long)row * 512;
  y[tid]       = f2bf((x0 - mean) * inv * g[tid] + b[tid]);
  y[tid + 256] = f2bf((x1 - mean) * inv * g[tid + 256] + b[tid + 256]);
}

// LayerNorm of row 0 only, fp32 out (final cls vector)
__global__ __launch_bounds__(256) void ln_row0_f32_k(const float* __restrict__ X,
    const float* __restrict__ g, const float* __restrict__ b, float* __restrict__ out) {
  const int tid = threadIdx.x;
  float x0 = X[tid], x1 = X[tid + 256];
  __shared__ float s1[256], s2[256];
  s1[tid] = x0 + x1; s2[tid] = x0 * x0 + x1 * x1;
  __syncthreads();
  for (int st = 128; st > 0; st >>= 1) {
    if (tid < st) { s1[tid] += s1[tid + st]; s2[tid] += s2[tid + st]; }
    __syncthreads();
  }
  float mean = s1[0] * (1.f / 512.f);
  float var  = s2[0] * (1.f / 512.f) - mean * mean;
  float inv  = rsqrtf(var + 1e-5f);
  out[tid]       = (x0 - mean) * inv * g[tid] + b[tid];
  out[tid + 256] = (x1 - mean) * inv * g[tid + 256] + b[tid + 256];
}

// ---------------------------------------------------------------------------
// Row softmax with logit scale; fp32 in, fp32 or bf16 out
// ---------------------------------------------------------------------------
__global__ __launch_bounds__(256) void softmax_rows_k(const float* __restrict__ X,
    long long ld, long long bs, int L, float scale,
    float* Of, long long ldo, long long bso, unsigned short* Ob) {
  const long long row = blockIdx.x, bat = blockIdx.y;
  const float* x = X + bat * bs + row * ld;
  const int tid = threadIdx.x;
  __shared__ float red[256];
  float m = -3.4e38f;
  for (int i = tid; i < L; i += 256) m = fmaxf(m, x[i] * scale);
  red[tid] = m; __syncthreads();
  for (int st = 128; st > 0; st >>= 1) { if (tid < st) red[tid] = fmaxf(red[tid], red[tid + st]); __syncthreads(); }
  m = red[0]; __syncthreads();
  float s = 0.f;
  for (int i = tid; i < L; i += 256) s += expf(x[i] * scale - m);
  red[tid] = s; __syncthreads();
  for (int st = 128; st > 0; st >>= 1) { if (tid < st) red[tid] += red[tid + st]; __syncthreads(); }
  float inv = 1.f / red[0];
  if (Of) {
    float* o = Of + bat * bso + row * ldo;
    for (int i = tid; i < L; i += 256) o[i] = expf(x[i] * scale - m) * inv;
  } else {
    unsigned short* o = Ob + bat * bso + row * ldo;
    for (int i = tid; i < L; i += 256) o[i] = f2bf(expf(x[i] * scale - m) * inv);
  }
}

// ---------------------------------------------------------------------------
// Landmark means over contiguous chunks of l tokens (q and k heads)
// grid (256, 8), block 64
// ---------------------------------------------------------------------------
__global__ void landmarks_k(const unsigned short* __restrict__ qkv, int n, int l,
                            unsigned short* __restrict__ QL, unsigned short* __restrict__ KL) {
  const int m = blockIdx.x, h = blockIdx.y, d = threadIdx.x;
  float aq = 0.f, ak = 0.f;
  long long base = ((long long)m * l) * 1536 + h * 64 + d;
  for (int j = 0; j < l; ++j) { aq += bf2f(qkv[base]); ak += bf2f(qkv[base + 512]); base += 1536; }
  float invl = 1.f / (float)l;
  long long o = ((long long)h * 256 + m) * 64 + d;
  QL[o] = f2bf(aq * invl); KL[o] = f2bf(ak * invl);
}

// ---------------------------------------------------------------------------
// Depthwise 1D conv (33 taps) along tokens per head, accumulate into OUT
// ---------------------------------------------------------------------------
__global__ __launch_bounds__(256) void conv_res_k(const unsigned short* __restrict__ qkv,
    const float* __restrict__ w, float* __restrict__ OUT, int n) {
  const int d = threadIdx.x & 63, sub = threadIdx.x >> 6;
  const int tok = blockIdx.x * 4 + sub, h = blockIdx.y;
  if (tok >= n) return;
  float acc = 0.f;
  for (int t = 0; t < 33; ++t) {
    int src = tok + t - 16;
    if (src >= 0 && src < n)
      acc += bf2f(qkv[(long long)src * 1536 + 1024 + h * 64 + d]) * w[h * 33 + t];
  }
  OUT[(long long)tok * 512 + h * 64 + d] += acc;
}

// ---------------------------------------------------------------------------
// PPEG: identity + depthwise 7x7 + 5x5 + 3x3 on (512, s, s)
// ---------------------------------------------------------------------------
__global__ void ppeg_k(const float* __restrict__ Hin, float* __restrict__ Hout,
    const float* __restrict__ w7, const float* __restrict__ b7,
    const float* __restrict__ w5, const float* __restrict__ b5,
    const float* __restrict__ w3, const float* __restrict__ b3, int s) {
  const long long total = (long long)s * s * 512;
  for (long long idx = (long long)blockIdx.x * 256 + threadIdx.x; idx < total;
       idx += (long long)gridDim.x * 256) {
    int c = (int)(idx & 511);
    long long sp = idx >> 9;
    int x = (int)(sp % s), y = (int)(sp / s);
    float acc = Hin[(sp + 1) * 512 + c] + b7[c] + b5[c] + b3[c];
    for (int ky = 0; ky < 7; ++ky) { int yy = y + ky - 3; if (yy < 0 || yy >= s) continue;
      for (int kx = 0; kx < 7; ++kx) { int xx = x + kx - 3; if (xx < 0 || xx >= s) continue;
        acc += Hin[((long long)yy * s + xx + 1) * 512 + c] * w7[c * 49 + ky * 7 + kx]; } }
    for (int ky = 0; ky < 5; ++ky) { int yy = y + ky - 2; if (yy < 0 || yy >= s) continue;
      for (int kx = 0; kx < 5; ++kx) { int xx = x + kx - 2; if (xx < 0 || xx >= s) continue;
        acc += Hin[((long long)yy * s + xx + 1) * 512 + c] * w5[c * 25 + ky * 5 + kx]; } }
    for (int ky = 0; ky < 3; ++ky) { int yy = y + ky - 1; if (yy < 0 || yy >= s) continue;
      for (int kx = 0; kx < 3; ++kx) { int xx = x + kx - 1; if (xx < 0 || xx >= s) continue;
        acc += Hin[((long long)yy * s + xx + 1) * 512 + c] * w3[c * 9 + ky * 3 + kx]; } }
    Hout[(sp + 1) * 512 + c] = acc;
  }
}

// ---------------------------------------------------------------------------
// Assemble token stream: [cls | feats | wraparound pad]
// ---------------------------------------------------------------------------
__global__ void assemble_k(const float* __restrict__ feats, const float* __restrict__ cls,
                           float* __restrict__ H, int N, int sq) {
  const long long total = (long long)(sq + 1) * 512;
  for (long long idx = (long long)blockIdx.x * 256 + threadIdx.x; idx < total;
       idx += (long long)gridDim.x * 256) {
    int c = (int)(idx & 511);
    long long t = idx >> 9;
    float v;
    if (t == 0) v = cls[c];
    else { long long f = t - 1; if (f >= N) f -= N; v = feats[f * 512 + c]; }
    H[idx] = v;
  }
}

// ---------------------------------------------------------------------------
// pinv init reductions
// ---------------------------------------------------------------------------
__global__ __launch_bounds__(256) void abs_row_sums_k(const float* __restrict__ A, float* __restrict__ R) {
  const int row = blockIdx.x, h = blockIdx.y, tid = threadIdx.x;
  __shared__ float red[256];
  red[tid] = fabsf(A[((long long)h * 256 + row) * 256 + tid]);
  __syncthreads();
  for (int st = 128; st > 0; st >>= 1) { if (tid < st) red[tid] += red[tid + st]; __syncthreads(); }
  if (tid == 0) R[h * 256 + row] = red[0];
}
__global__ __launch_bounds__(256) void abs_col_sums_k(const float* __restrict__ A, float* __restrict__ C) {
  const int col = blockIdx.x, h = blockIdx.y, tid = threadIdx.x;
  __shared__ float red[256];
  red[tid] = fabsf(A[(long long)h * 65536 + (long long)tid * 256 + col]);
  __syncthreads();
  for (int st = 128; st > 0; st >>= 1) { if (tid < st) red[tid] += red[tid + st]; __syncthreads(); }
  if (tid == 0) C[h * 256 + col] = red[0];
}
__global__ __launch_bounds__(256) void pinv_scale_k(const float* __restrict__ R, const float* __restrict__ C,
                                                    float* __restrict__ S) {
  const int tid = threadIdx.x;
  __shared__ float r1[256], r2[256];
  float mr = -3.4e38f, mc = -3.4e38f;
  for (int i = tid; i < 2048; i += 256) { mr = fmaxf(mr, R[i]); mc = fmaxf(mc, C[i]); }
  r1[tid] = mr; r2[tid] = mc; __syncthreads();
  for (int st = 128; st > 0; st >>= 1) {
    if (tid < st) { r1[tid] = fmaxf(r1[tid], r1[tid + st]); r2[tid] = fmaxf(r2[tid], r2[tid + st]); }
    __syncthreads();
  }
  if (tid == 0) S[0] = 1.f / (r1[0] * r2[0]);
}
__global__ void zinit_k(const float* __restrict__ A, const float* __restrict__ S, float* __restrict__ Z) {
  const float sc = S[0];
  for (long long idx = (long long)blockIdx.x * 256 + threadIdx.x; idx < 8LL * 65536;
       idx += (long long)gridDim.x * 256) {
    long long h = idx >> 16; int r = (int)((idx >> 8) & 255), c = (int)(idx & 255);
    Z[idx] = A[h * 65536 + (long long)c * 256 + r] * sc;   // transpose * scale
  }
}

// ---------------------------------------------------------------------------
// Gated-attention pooling head + hazard head (tiny; one block)
// ---------------------------------------------------------------------------
__global__ __launch_bounds__(256) void head_k(const float* __restrict__ X,
    const float* __restrict__ aaw, const float* __restrict__ aab,
    const float* __restrict__ abw, const float* __restrict__ abb,
    const float* __restrict__ acw, const float* __restrict__ acb,
    const float* __restrict__ f2w, const float* __restrict__ f2b,
    float* __restrict__ out) {
  __shared__ float Am[2][512];
  __shared__ float red[256];
  __shared__ float sc[2];
  __shared__ float hp[512];
  const int tid = threadIdx.x;
  for (int r = 0; r < 2; ++r)
    for (int c = tid; c < 512; c += 256) {
      float sa = aab[c], sg = abb[c];
      for (int k = 0; k < 512; ++k) {
        float xv = X[r * 512 + k];
        sa += xv * aaw[(long long)k * 512 + c];
        sg += xv * abw[(long long)k * 512 + c];
      }
      Am[r][c] = tanhf(sa) * (1.f / (1.f + expf(-sg)));
    }
  __syncthreads();
  for (int r = 0; r < 2; ++r) {
    float s = 0.f;
    for (int c = tid; c < 512; c += 256) { float v = Am[r][c]; s += v * v; }
    red[tid] = s; __syncthreads();
    for (int st = 128; st > 0; st >>= 1) { if (tid < st) red[tid] += red[tid + st]; __syncthreads(); }
    float nrm = fmaxf(sqrtf(red[0]), 1e-12f);
    __syncthreads();
    float ds = 0.f;
    for (int c = tid; c < 512; c += 256) ds += (Am[r][c] / nrm) * acw[c];
    red[tid] = ds; __syncthreads();
    for (int st = 128; st > 0; st >>= 1) { if (tid < st) red[tid] += red[tid + st]; __syncthreads(); }
    if (tid == 0) sc[r] = red[0] + acb[0];
    __syncthreads();
  }
  float m = fmaxf(sc[0], sc[1]);
  float e0 = expf(sc[0] - m), e1 = expf(sc[1] - m);
  float w0 = e0 / (e0 + e1), w1 = e1 / (e0 + e1);
  for (int c = tid; c < 512; c += 256) hp[c] = w0 * X[c] + w1 * X[512 + c];
  __syncthreads();
  if (tid < 4) {
    float s = f2b[tid];
    for (int k = 0; k < 512; ++k) s += hp[k] * f2w[(long long)k * 4 + tid];
    out[tid] = s;                           // logits
    out[4 + tid] = 1.f / (1.f + expf(-s));  // hazards
  }
  __syncthreads();
  if (tid == 0) {
    float sp = 1.f;
    for (int j = 0; j < 4; ++j) { sp *= (1.f - out[4 + j]); out[8 + j] = sp; }  // S = cumprod(1-h)
  }
}

// ===========================================================================
// Host orchestration
// ===========================================================================
static inline int cdiv(long long a, long long b) { return (int)((a + b - 1) / b); }

struct Bufs {
  unsigned short *fc1wB, *qkvwB[2], *outwB[2];
  unsigned short *HB, *XPAD, *QKV, *BIG2, *OUTB, *QL, *KL, *ZOUTB;
  float *FEATS, *HTOK, *HTOK2, *BIG1, *OUT;
  float *A2, *XZ, *T, *U, *Z0, *Z1, *A3V, *ZOUT, *ROWS, *COLS, *SCALE, *XCLS;
};
struct Prm {
  const float *fc1_w,*fc1_b,*cls_tok,*normf_g,*normf_b,*fc2_w,*fc2_b;
  const float *aa_w,*aa_b,*ab_w,*ab_b,*ac_w,*ac_b;
  const float *c7w,*c7b,*c5w,*c5b,*c3w,*c3b;
  const float *ln_g[2],*ln_b[2],*out_b[2],*res_k[2];
};

static void launch_gemm_bf(const GemmBF& p, int batch, hipStream_t st) {
  dim3 g(cdiv(p.N, 128), cdiv(p.M, 128), batch);
  gemm_bf16_k<<<g, 256, 0, st>>>(p);
}
static void launch_gemm_f32(const GemmF32& p, int batch, hipStream_t st) {
  dim3 g(cdiv(p.N, 64), cdiv(p.M, 64), batch);
  gemm_f32_k<<<g, 256, 0, st>>>(p);
}

static void attn_block(float* stream_buf, int ntok, int n, int pad, int l,
                       const unsigned short* qkvwB, const unsigned short* outwB,
                       const float* out_b, const float* res_k,
                       const float* ln_g, const float* ln_b,
                       Bufs& B, hipStream_t st) {
  // LN -> bf16, front-padded with zeros
  if (pad > 0) hipMemsetAsync(B.XPAD, 0, (size_t)pad * 512 * 2, st);
  layernorm_bf16_k<<<ntok, 256, 0, st>>>(stream_buf, ln_g, ln_b, B.XPAD + (long long)pad * 512);

  // qkv = x @ Wqkv   (n x 512 x 1536) -> bf16
  { GemmBF p{}; p.A=B.XPAD; p.lda=512; p.sA=0; p.B=qkvwB; p.ldb=1536; p.sB=0; p.transB=0;
    p.bias=nullptr; p.relu=0; p.Cf=nullptr; p.Cb=B.QKV; p.ldc=1536; p.sC=0;
    p.Cres=nullptr; p.ldres=0; p.rowOff=0; p.M=n; p.N=1536; p.K=512;
    launch_gemm_bf(p, 1, st); }

  // landmarks ql, kl (8 x 256 x 64)
  landmarks_k<<<dim3(256, 8), 64, 0, st>>>(B.QKV, n, l, B.QL, B.KL);

  // a2 logits = ql @ kl^T  -> fp32, softmax in place
  { GemmBF p{}; p.A=B.QL; p.lda=64; p.sA=256*64; p.B=B.KL; p.ldb=64; p.sB=256*64; p.transB=1;
    p.bias=nullptr; p.relu=0; p.Cf=B.A2; p.Cb=nullptr; p.ldc=256; p.sC=65536;
    p.Cres=nullptr; p.ldres=0; p.rowOff=0; p.M=256; p.N=256; p.K=64;
    launch_gemm_bf(p, 8, st); }
  softmax_rows_k<<<dim3(256, 8), 256, 0, st>>>(B.A2, 256, 65536, 256, 0.125f,
                                               B.A2, 256, 65536, nullptr);

  // Moore-Penrose pinv of a2 (fp32, V_WMMA_F32_16X16X4_F32)
  abs_row_sums_k<<<dim3(256, 8), 256, 0, st>>>(B.A2, B.ROWS);
  abs_col_sums_k<<<dim3(256, 8), 256, 0, st>>>(B.A2, B.COLS);
  pinv_scale_k<<<1, 256, 0, st>>>(B.ROWS, B.COLS, B.SCALE);
  zinit_k<<<2048, 256, 0, st>>>(B.A2, B.SCALE, B.Z0);
  float* z = B.Z0; float* znew = B.Z1;
  for (int it = 0; it < 6; ++it) {
    GemmF32 p{};
    p.lda=p.ldb=p.ldc=p.ldm=256; p.sA=p.sB=p.sC=p.sM=65536;
    // XZ = a2 @ z
    p.A=B.A2; p.B=z; p.C=B.XZ; p.Madd=nullptr; p.alpha=1.f; p.beta=0.f; p.M=p.N=p.K=256;
    launch_gemm_f32(p, 8, st);
    // T = 7*XZ - XZ@XZ            ( = xz @ (7I - xz) )
    p.A=B.XZ; p.B=B.XZ; p.C=B.T; p.Madd=B.XZ; p.alpha=-1.f; p.beta=7.f;
    launch_gemm_f32(p, 8, st);
    // U = 15*XZ - XZ@T            ( = xz @ (15I - T) )
    p.A=B.XZ; p.B=B.T; p.C=B.U; p.Madd=B.XZ; p.alpha=-1.f; p.beta=15.f;
    launch_gemm_f32(p, 8, st);
    // z' = 0.25*(13 z - z@U)      ( = 0.25 z @ (13I - U) )
    p.A=z; p.B=B.U; p.C=znew; p.Madd=z; p.alpha=-0.25f; p.beta=3.25f;
    launch_gemm_f32(p, 8, st);
    float* t = z; z = znew; znew = t;
  }

  // a3 logits = ql @ k^T (256 x n), softmax -> bf16
  { GemmBF p{}; p.A=B.QL; p.lda=64; p.sA=256*64; p.B=B.QKV+512; p.ldb=1536; p.sB=64; p.transB=1;
    p.bias=nullptr; p.relu=0; p.Cf=B.BIG1; p.Cb=nullptr; p.ldc=n; p.sC=(long long)256*n;
    p.Cres=nullptr; p.ldres=0; p.rowOff=0; p.M=256; p.N=n; p.K=64;
    launch_gemm_bf(p, 8, st); }
  softmax_rows_k<<<dim3(256, 8), 256, 0, st>>>(B.BIG1, n, (long long)256*n, n, 0.125f,
                                               nullptr, n, (long long)256*n, B.BIG2);

  // a3v = a3 @ v  (256 x 64, K=n) -> fp32
  { GemmBF p{}; p.A=B.BIG2; p.lda=n; p.sA=(long long)256*n; p.B=B.QKV+1024; p.ldb=1536; p.sB=64; p.transB=0;
    p.bias=nullptr; p.relu=0; p.Cf=B.A3V; p.Cb=nullptr; p.ldc=64; p.sC=256*64;
    p.Cres=nullptr; p.ldres=0; p.rowOff=0; p.M=256; p.N=64; p.K=n;
    launch_gemm_bf(p, 8, st); }

  // zout = z @ a3v (fp32 WMMA), then -> bf16
  { GemmF32 p{}; p.A=z; p.lda=256; p.sA=65536; p.B=B.A3V; p.ldb=64; p.sB=256*64;
    p.C=B.ZOUT; p.ldc=64; p.sC=256*64; p.Madd=nullptr; p.ldm=0; p.sM=0;
    p.alpha=1.f; p.beta=0.f; p.M=256; p.N=64; p.K=256;
    launch_gemm_f32(p, 8, st); }
  f32_to_bf16_k<<<512, 256, 0, st>>>(B.ZOUT, B.ZOUTB, 8LL * 256 * 64);

  // a1 logits = q @ kl^T (n x 256), softmax -> bf16
  { GemmBF p{}; p.A=B.QKV; p.lda=1536; p.sA=64; p.B=B.KL; p.ldb=64; p.sB=256*64; p.transB=1;
    p.bias=nullptr; p.relu=0; p.Cf=B.BIG1; p.Cb=nullptr; p.ldc=256; p.sC=(long long)n*256;
    p.Cres=nullptr; p.ldres=0; p.rowOff=0; p.M=n; p.N=256; p.K=64;
    launch_gemm_bf(p, 8, st); }
  softmax_rows_k<<<dim3(n, 8), 256, 0, st>>>(B.BIG1, 256, (long long)n*256, 256, 0.125f,
                                             nullptr, 256, (long long)n*256, B.BIG2);

  // out = a1 @ zout  (n x 64 per head -> n x 512 head-concat, fp32)
  { GemmBF p{}; p.A=B.BIG2; p.lda=256; p.sA=(long long)n*256; p.B=B.ZOUTB; p.ldb=64; p.sB=256*64; p.transB=0;
    p.bias=nullptr; p.relu=0; p.Cf=B.OUT; p.Cb=nullptr; p.ldc=512; p.sC=64;
    p.Cres=nullptr; p.ldres=0; p.rowOff=0; p.M=n; p.N=64; p.K=256;
    launch_gemm_bf(p, 8, st); }

  // out += depthwise conv1d(v)
  conv_res_k<<<dim3(cdiv(n, 4), 8), 256, 0, st>>>(B.QKV, res_k, B.OUT, n);

  // project + residual add into stream (dropping the front pad rows)
  f32_to_bf16_k<<<4096, 256, 0, st>>>(B.OUT, B.OUTB, (long long)n * 512);
  { GemmBF p{}; p.A=B.OUTB; p.lda=512; p.sA=0; p.B=outwB; p.ldb=512; p.sB=0; p.transB=0;
    p.bias=out_b; p.relu=0; p.Cf=stream_buf; p.Cb=nullptr; p.ldc=512; p.sC=0;
    p.Cres=stream_buf; p.ldres=512; p.rowOff=pad; p.M=n; p.N=512; p.K=512;
    launch_gemm_bf(p, 1, st); }
}

extern "C" void kernel_launch(void* const* d_in, const int* in_sizes, int n_in,
                              void* d_out, int out_size, void* d_ws, size_t ws_size,
                              hipStream_t stream) {
  (void)n_in; (void)out_size; (void)ws_size;
  const float* h1 = (const float*)d_in[0];
  const float* h2 = (const float*)d_in[1];
  const int N1 = in_sizes[0] / 1024, N2 = in_sizes[1] / 1024;

  // --- resolve params (insertion-ordered dict vs. sorted pytree flatten) ---
  Prm P{};
  const float* qkv_w[2]; const float* out_w[2];
  auto gi = [&](int i)->const float* { return (const float*)d_in[i]; };
  if (in_sizes[2] == 1024 * 512) {  // insertion order: fc1_w first
    P.fc1_w=gi(2);  P.fc1_b=gi(3);  P.cls_tok=gi(4); P.normf_g=gi(5); P.normf_b=gi(6);
    P.fc2_w=gi(7);  P.fc2_b=gi(8);
    P.aa_w=gi(9);   P.aa_b=gi(10);  P.ab_w=gi(11);   P.ab_b=gi(12);
    P.ac_w=gi(13);  P.ac_b=gi(14);
    P.c7w=gi(15);   P.c7b=gi(16);   P.c5w=gi(17);    P.c5b=gi(18); P.c3w=gi(19); P.c3b=gi(20);
    P.ln_g[0]=gi(21); P.ln_b[0]=gi(22); qkv_w[0]=gi(23); out_w[0]=gi(24); P.out_b[0]=gi(25); P.res_k[0]=gi(26);
    P.ln_g[1]=gi(27); P.ln_b[1]=gi(28); qkv_w[1]=gi(29); out_w[1]=gi(30); P.out_b[1]=gi(31); P.res_k[1]=gi(32);
  } else {                          // alphabetically sorted keys
    P.aa_b=gi(2);   P.aa_w=gi(3);   P.ab_b=gi(4);    P.ab_w=gi(5);
    P.ac_b=gi(6);   P.ac_w=gi(7);   P.cls_tok=gi(8);
    P.c3b=gi(9);    P.c3w=gi(10);   P.c5b=gi(11);    P.c5w=gi(12); P.c7b=gi(13); P.c7w=gi(14);
    P.fc1_b=gi(15); P.fc1_w=gi(16); P.fc2_b=gi(17);  P.fc2_w=gi(18);
    P.ln_b[0]=gi(19); P.ln_g[0]=gi(20); P.ln_b[1]=gi(21); P.ln_g[1]=gi(22);
    P.normf_b=gi(23); P.normf_g=gi(24);
    P.out_b[0]=gi(25); out_w[0]=gi(26); P.out_b[1]=gi(27); out_w[1]=gi(28);
    qkv_w[0]=gi(29);  qkv_w[1]=gi(30); P.res_k[0]=gi(31); P.res_k[1]=gi(32);
  }

  // --- geometry ---
  auto geom = [](int N, int& s, int& sq, int& ntok, int& pad, int& n, int& l) {
    s = 1; while (s * s < N) ++s;
    sq = s * s; ntok = sq + 1;
    pad = (256 - (ntok % 256)) % 256; n = ntok + pad; l = n / 256;
  };
  int s1,sq1,ntok1,pad1,n1,l1, s2,sq2,ntok2,pad2,n2,l2;
  geom(N1, s1, sq1, ntok1, pad1, n1, l1);
  geom(N2, s2, sq2, ntok2, pad2, n2, l2);
  const int Nmax = (N1 > N2) ? N1 : N2;
  const int nmax = (n1 > n2) ? n1 : n2;
  const int ntokmax = (ntok1 > ntok2) ? ntok1 : ntok2;

  // --- workspace carve (bump allocator, 256B aligned, with aliasing) ---
  char* base = (char*)d_ws; size_t off = 0;
  auto carve = [&](size_t bytes)->void* {
    off = (off + 255) & ~(size_t)255; void* p = base + off; off += bytes; return p;
  };
  Bufs B{};
  B.fc1wB    = (unsigned short*)carve(1024LL*512*2);
  B.qkvwB[0] = (unsigned short*)carve(512LL*1536*2);
  B.qkvwB[1] = (unsigned short*)carve(512LL*1536*2);
  B.outwB[0] = (unsigned short*)carve(512LL*512*2);
  B.outwB[1] = (unsigned short*)carve(512LL*512*2);
  {  // QKV (bf16, n x 1536) aliases HB (input bf16, N x 1024): disjoint lifetimes
    size_t qkvB = (size_t)nmax*1536*2, hbB = (size_t)Nmax*1024*2;
    void* p = carve(qkvB > hbB ? qkvB : hbB);
    B.QKV = (unsigned short*)p; B.HB = (unsigned short*)p;
  }
  {  // BIG1 (fp32 logits 8x256xn) aliases FEATS (fp32 N x 512)
    size_t bB = 8ULL*256*nmax*4, fB = (size_t)Nmax*512*4;
    void* p = carve(bB > fB ? bB : fB);
    B.BIG1 = (float*)p; B.FEATS = (float*)p;
  }
  B.BIG2  = (unsigned short*)carve(8ULL*256*nmax*2);
  B.HTOK  = (float*)carve((size_t)ntokmax*512*4);
  B.HTOK2 = (float*)carve((size_t)ntokmax*512*4);
  B.XPAD  = (unsigned short*)carve((size_t)nmax*512*2);
  B.OUT   = (float*)carve((size_t)nmax*512*4);
  B.OUTB  = (unsigned short*)carve((size_t)nmax*512*2);
  B.QL    = (unsigned short*)carve(8ULL*256*64*2);
  B.KL    = (unsigned short*)carve(8ULL*256*64*2);
  B.A2    = (float*)carve(8ULL*65536*4);
  B.XZ    = (float*)carve(8ULL*65536*4);
  B.T     = (float*)carve(8ULL*65536*4);
  B.U     = (float*)carve(8ULL*65536*4);
  B.Z0    = (float*)carve(8ULL*65536*4);
  B.Z1    = (float*)carve(8ULL*65536*4);
  B.A3V   = (float*)carve(8ULL*256*64*4);
  B.ZOUT  = (float*)carve(8ULL*256*64*4);
  B.ZOUTB = (unsigned short*)carve(8ULL*256*64*2);
  B.ROWS  = (float*)carve(2048*4);
  B.COLS  = (float*)carve(2048*4);
  B.SCALE = (float*)carve(256);
  B.XCLS  = (float*)carve(2*512*4);

  // --- convert weights to bf16 (recomputed every call: deterministic) ---
  f32_to_bf16_k<<<2048, 256, 0, stream>>>(P.fc1_w, B.fc1wB, 1024LL*512);
  f32_to_bf16_k<<<2048, 256, 0, stream>>>(qkv_w[0], B.qkvwB[0], 512LL*1536);
  f32_to_bf16_k<<<2048, 256, 0, stream>>>(qkv_w[1], B.qkvwB[1], 512LL*1536);
  f32_to_bf16_k<<<1024, 256, 0, stream>>>(out_w[0], B.outwB[0], 512LL*512);
  f32_to_bf16_k<<<1024, 256, 0, stream>>>(out_w[1], B.outwB[1], 512LL*512);

  // --- two branches ---
  for (int br = 0; br < 2; ++br) {
    const float* hin = br ? h2 : h1;
    const int N    = br ? N2 : N1;
    const int s    = br ? s2 : s1;
    const int sq   = br ? sq2 : sq1;
    const int ntok = br ? ntok2 : ntok1;
    const int pad  = br ? pad2 : pad1;
    const int n    = br ? n2 : n1;
    const int l    = br ? l2 : l1;

    f32_to_bf16_k<<<8192, 256, 0, stream>>>(hin, B.HB, (long long)N * 1024);
    { GemmBF p{}; p.A=B.HB; p.lda=1024; p.sA=0; p.B=B.fc1wB; p.ldb=512; p.sB=0; p.transB=0;
      p.bias=P.fc1_b; p.relu=1; p.Cf=B.FEATS; p.Cb=nullptr; p.ldc=512; p.sC=0;
      p.Cres=nullptr; p.ldres=0; p.rowOff=0; p.M=N; p.N=512; p.K=1024;
      launch_gemm_bf(p, 1, stream); }
    assemble_k<<<cdiv((long long)(sq + 1) * 512, 256), 256, 0, stream>>>(
        B.FEATS, P.cls_tok, B.HTOK, N, sq);

    // attention block 1 (residual into HTOK)
    attn_block(B.HTOK, ntok, n, pad, l, B.qkvwB[0], B.outwB[0],
               P.out_b[0], P.res_k[0], P.ln_g[0], P.ln_b[0], B, stream);

    // PPEG (HTOK -> HTOK2)
    copy_f32_k<<<2, 256, 0, stream>>>(B.HTOK, B.HTOK2, 512);   // cls row
    ppeg_k<<<cdiv((long long)sq * 512, 256), 256, 0, stream>>>(
        B.HTOK, B.HTOK2, P.c7w, P.c7b, P.c5w, P.c5b, P.c3w, P.c3b, s);

    // attention block 2 (residual into HTOK2)
    attn_block(B.HTOK2, ntok, n, pad, l, B.qkvwB[1], B.outwB[1],
               P.out_b[1], P.res_k[1], P.ln_g[1], P.ln_b[1], B, stream);

    // final LN, cls token -> XCLS row
    ln_row0_f32_k<<<1, 256, 0, stream>>>(B.HTOK2, P.normf_g, P.normf_b, B.XCLS + br * 512);
  }

  // --- pooling + survival head -> d_out (logits | hazards | S : 12 floats) ---
  head_k<<<1, 256, 0, stream>>>(B.XCLS, P.aa_w, P.aa_b, P.ab_w, P.ab_b,
                                P.ac_w, P.ac_b, P.fc2_w, P.fc2_b, (float*)d_out);
}
